// SimpleGNNBlock_8203387535569
// MI455X (gfx1250) — compile-verified, matrix operand order checked by reading, not run on hardware
//
#include <hip/hip_runtime.h>
#include <hip/hip_bf16.h>

// ---------------------------------------------------------------------------
// SimpleGNNBlock for MI455X (gfx1250, wave32).
// GEMMs: v_wmma_f32_16x16x32_f16 (f16 A/B, f32 acc), B panel staged in LDS
//        via GLOBAL_LOAD_ASYNC_TO_LDS_B128 (ASYNCcnt) and read as ds_load_b128.
// Edge aggregation: wave-per-edge float4 gather + global_atomic_add_f32.
// Segment-max for GAT softmax: monotone float->u32 key + atomicMax.
// LayerNorm/SiLU: wave32 shuffle reductions, one wave per node.
// Workspace requirement: ~181 MB.
// ---------------------------------------------------------------------------

#define USE_ASYNC_LDS 1   // 1: global_load_async_to_lds_b128 staging; 0: ds_store staging

typedef __attribute__((ext_vector_type(16))) _Float16 v16h;
typedef __attribute__((ext_vector_type(8)))  _Float16 v8h;
typedef __attribute__((ext_vector_type(8)))  float    v8f;

#define GNN_N     50000
#define GNN_D     128
#define GNN_E     800000
#define GNN_EA    (GNN_E + GNN_N)   // edges + self loops
#define GNN_NEG   0.2f
#define GNN_EPS   1e-5f

// ---- monotone float <-> uint key (for atomic max on signed floats) --------
__device__ __forceinline__ unsigned fkey(float x) {
  unsigned b = __float_as_uint(x);
  return (b & 0x80000000u) ? ~b : (b | 0x80000000u);
}
__device__ __forceinline__ float fkeyinv(unsigned k) {
  unsigned b = (k & 0x80000000u) ? (k & 0x7FFFFFFFu) : ~k;
  return __uint_as_float(b);
}
#define FKEY_NEG_INF 0x007FFFFFu   // fkey(-inf)

// ---------------------------------------------------------------------------
// utility fills / conversions
// ---------------------------------------------------------------------------
__global__ void k_fill_f32(float* __restrict__ p, float v, int n) {
  int i = blockIdx.x * blockDim.x + threadIdx.x;
  if (i < n) p[i] = v;
}
__global__ void k_fill_u32(unsigned* __restrict__ p, unsigned v, int n) {
  int i = blockIdx.x * blockDim.x + threadIdx.x;
  if (i < n) p[i] = v;
}
__global__ void k_cvt_f16(const float* __restrict__ x, _Float16* __restrict__ y, int n) {
  int i = blockIdx.x * blockDim.x + threadIdx.x;
  if (i < n) y[i] = (_Float16)x[i];
}
// W[K x Ncols] row-major  ->  Wt[Ncols x K] f16 (each output column contiguous over K)
__global__ void k_cvt_wT(const float* __restrict__ W, _Float16* __restrict__ Wt,
                         int K, int Ncols) {
  int i = blockIdx.x * blockDim.x + threadIdx.x;
  if (i >= K * Ncols) return;
  int n = i / K, k = i - n * K;
  Wt[(size_t)n * K + k] = (_Float16)W[(size_t)k * Ncols + n];
}

// ---------------------------------------------------------------------------
// GEMM:  C[M x 16*NT] = A[M x 128] (f16) @ B (f16, Bt stored [cols][128]).
// One wave per 16-row tile; WMMA f32_16x16x32_f16, K = 128 = 4 k-steps.
// The whole Bt panel (NT*16 cols x 128 k, 32/64 KB) is staged into LDS once
// per workgroup (4 waves reuse it), then read back as ds_load_b128.
// A-fragment layout (ISA 7.12.2, 16-bit A 16x32): lane M = lane&15,
// K-half select = lane>>4: elems 0..7 -> K = 8*(lane>>4)+0..7,
//                          elems 8..15 -> K = 16 + 8*(lane>>4)+0..7.
// ---------------------------------------------------------------------------
template<int NT>
__global__ void k_gemm_f16_lds(const _Float16* __restrict__ A,
                               const _Float16* __restrict__ Bt,
                               float* __restrict__ C, int Mtiles) {
  __shared__ _Float16 bs[NT * 16 * 128];          // 32 KB (NT=8) / 64 KB (NT=16)
  const int K   = 128;
  const int ldc = NT << 4;
  const int tid = threadIdx.x;

  // ---- stage Bt panel into LDS (uniform trip count: no divergence) --------
#if USE_ASYNC_LDS
  {
    const int totalBytes = NT * 16 * 128 * 2;
    unsigned lbase = (unsigned)(uintptr_t)bs;     // LDS byte offset of bs
    unsigned long long gbase = (unsigned long long)(const void*)Bt;
    for (int o = tid * 16; o < totalBytes; o += (int)blockDim.x * 16) {
      unsigned           laddr = lbase + (unsigned)o;
      unsigned long long gaddr = gbase + (unsigned long long)o;
      asm volatile("global_load_async_to_lds_b128 %0, %1, off"
                   :: "v"(laddr), "v"(gaddr) : "memory");
    }
    asm volatile("s_wait_asynccnt 0" ::: "memory");
  }
#else
  for (int o = tid * 8; o < NT * 16 * 128; o += (int)blockDim.x * 8)
    *(v8h*)(bs + o) = *(const v8h*)(Bt + o);
#endif
  __syncthreads();

  int wave = (blockIdx.x * blockDim.x + tid) >> 5;
  int lane = tid & 31;
  if (wave >= Mtiles) return;          // wave-uniform: EXEC all-ones for WMMA

  int row0 = wave << 4;
  int m  = lane & 15;
  int kh = (lane >> 4) << 3;           // 0 or 8

  // ---- A fragments for all 4 k-steps (streamed once from HBM) -------------
  v16h afrag[4];
  const _Float16* pa = A + (size_t)(row0 + m) * K + kh;
#pragma unroll
  for (int kt = 0; kt < 4; ++kt) {
    v8h lo = *(const v8h*)(pa + kt * 32);
    v8h hi = *(const v8h*)(pa + kt * 32 + 16);
    afrag[kt] = __builtin_shufflevector(lo, hi,
        0,1,2,3,4,5,6,7,8,9,10,11,12,13,14,15);
  }

  // ---- sweep column tiles, B from LDS -------------------------------------
  for (int nt = 0; nt < NT; ++nt) {
    v8f acc = {0.f,0.f,0.f,0.f,0.f,0.f,0.f,0.f};
    const _Float16* pb = bs + ((nt << 4) + m) * K + kh;
#pragma unroll
    for (int kt = 0; kt < 4; ++kt) {
      v8h lo = *(const v8h*)(pb + kt * 32);
      v8h hi = *(const v8h*)(pb + kt * 32 + 16);
      v16h bfrag = __builtin_shufflevector(lo, hi,
          0,1,2,3,4,5,6,7,8,9,10,11,12,13,14,15);
      acc = __builtin_amdgcn_wmma_f32_16x16x32_f16(
          false, afrag[kt], false, bfrag, (short)0, acc, false, false);
    }
    // D layout: elem r -> M = row0 + r + 8*(lane>>4), N = nt*16 + (lane&15)
    float* pc = C + (size_t)(row0 + ((lane >> 4) << 3)) * ldc + (nt << 4) + m;
#pragma unroll
    for (int r = 0; r < 8; ++r) pc[(size_t)r * ldc] = acc[r];
  }
}

// ---------------------------------------------------------------------------
// GCN pieces
// ---------------------------------------------------------------------------
__global__ void k_deg(const long long* __restrict__ dst, float* __restrict__ deg, int E) {
  int e = blockIdx.x * blockDim.x + threadIdx.x;
  if (e < E) atomicAdd(&deg[(int)dst[e]], 1.0f);
}
__global__ void k_rsqrt_inplace(float* __restrict__ d, int n) {
  int i = blockIdx.x * blockDim.x + threadIdx.x;
  if (i < n) d[i] = rsqrtf(d[i]);       // deg >= 1 (self loops) so no guard needed
}
// self-loop contribution + bias:  g1[n][c] = gcn_b[c] + h1[n][c]*dinv[n]^2
__global__ void k_gcn_self(const float* __restrict__ h1, const float* __restrict__ dinv,
                           const float* __restrict__ bias, float* __restrict__ g1, int n) {
  int i = blockIdx.x * blockDim.x + threadIdx.x;
  if (i >= n) return;
  int node = i >> 7, c = i & 127;
  float di = dinv[node];
  g1[i] = bias[c] + h1[i] * di * di;
}
// wave per edge: g1[dst] += h1[src] * dinv[src]*dinv[dst]
__global__ void k_gcn_edges(const long long* __restrict__ src, const long long* __restrict__ dst,
                            const float* __restrict__ h1, const float* __restrict__ dinv,
                            float* __restrict__ g1, int E) {
  int wid  = (blockIdx.x * blockDim.x + threadIdx.x) >> 5;
  int lane = threadIdx.x & 31;
  if (wid >= E) return;
  int s = (int)src[wid], d = (int)dst[wid];
  float nrm = dinv[s] * dinv[d];
  const float* ps = h1 + (size_t)s * GNN_D;
  float*       pd = g1 + (size_t)d * GNN_D;
  __builtin_prefetch(pd, 0, 1);                         // global_prefetch_b8
  float4 v = ((const float4*)ps)[lane];
  int c = lane << 2;
  atomicAdd(&pd[c + 0], v.x * nrm);
  atomicAdd(&pd[c + 1], v.y * nrm);
  atomicAdd(&pd[c + 2], v.z * nrm);
  atomicAdd(&pd[c + 3], v.w * nrm);
}

// ---------------------------------------------------------------------------
// LayerNorm + SiLU helper (wave32, one wave per 128-dim node row)
// ---------------------------------------------------------------------------
__device__ __forceinline__ float4 ln_silu4(float4 v, float4 g, float4 bt) {
  float s = v.x + v.y + v.z + v.w;
  float q = v.x*v.x + v.y*v.y + v.z*v.z + v.w*v.w;
#pragma unroll
  for (int o = 16; o; o >>= 1) {
    s += __shfl_xor(s, o, 32);
    q += __shfl_xor(q, o, 32);
  }
  float mu   = s * (1.0f / 128.0f);
  float var  = q * (1.0f / 128.0f) - mu * mu;
  float rinv = rsqrtf(var + GNN_EPS);
  float4 y;
  y.x = (v.x - mu) * rinv * g.x + bt.x;  y.x = y.x / (1.0f + __expf(-y.x));
  y.y = (v.y - mu) * rinv * g.y + bt.y;  y.y = y.y / (1.0f + __expf(-y.y));
  y.z = (v.z - mu) * rinv * g.z + bt.z;  y.z = y.z / (1.0f + __expf(-y.z));
  y.w = (v.w - mu) * rinv * g.w + bt.w;  y.w = y.w / (1.0f + __expf(-y.w));
  return y;
}

__global__ void k_ln_silu_f16(const float* __restrict__ in,
                              const float* __restrict__ gamma, const float* __restrict__ beta,
                              _Float16* __restrict__ out, int N) {
  int wid  = (blockIdx.x * blockDim.x + threadIdx.x) >> 5;
  int lane = threadIdx.x & 31;
  if (wid >= N) return;
  float4 v  = ((const float4*)(in + (size_t)wid * GNN_D))[lane];
  float4 g  = ((const float4*)gamma)[lane];
  float4 bt = ((const float4*)beta)[lane];
  float4 y  = ln_silu4(v, g, bt);
  _Float16* o = out + (size_t)wid * GNN_D + (lane << 2);
  o[0] = (_Float16)y.x; o[1] = (_Float16)y.y; o[2] = (_Float16)y.z; o[3] = (_Float16)y.w;
}

// ---------------------------------------------------------------------------
// GAT pieces
// ---------------------------------------------------------------------------
// per-node attention scalars: as_[n][h] = <h3[n][h][:], att_src[h][:]>, same for dst
__global__ void k_att(const float* __restrict__ h3,
                      const float* __restrict__ att_src, const float* __restrict__ att_dst,
                      float* __restrict__ as_, float* __restrict__ ad_, int N) {
  int wid  = (blockIdx.x * blockDim.x + threadIdx.x) >> 5;
  int lane = threadIdx.x & 31;
  if (wid >= N) return;
  float4 h0 = ((const float4*)(h3 + (size_t)wid * 256))[lane];
  float4 h1 = ((const float4*)(h3 + (size_t)wid * 256 + 128))[lane];
  float4 s0 = ((const float4*)att_src)[lane];
  float4 s1 = ((const float4*)(att_src + 128))[lane];
  float4 d0 = ((const float4*)att_dst)[lane];
  float4 d1 = ((const float4*)(att_dst + 128))[lane];
  float r0 = h0.x*s0.x + h0.y*s0.y + h0.z*s0.z + h0.w*s0.w;
  float r1 = h1.x*s1.x + h1.y*s1.y + h1.z*s1.z + h1.w*s1.w;
  float r2 = h0.x*d0.x + h0.y*d0.y + h0.z*d0.z + h0.w*d0.w;
  float r3 = h1.x*d1.x + h1.y*d1.y + h1.z*d1.z + h1.w*d1.w;
#pragma unroll
  for (int o = 16; o; o >>= 1) {
    r0 += __shfl_xor(r0, o, 32); r1 += __shfl_xor(r1, o, 32);
    r2 += __shfl_xor(r2, o, 32); r3 += __shfl_xor(r3, o, 32);
  }
  if (lane == 0) {
    as_[wid * 2]     = r0; as_[wid * 2 + 1] = r1;
    ad_[wid * 2]     = r2; ad_[wid * 2 + 1] = r3;
  }
}

__device__ __forceinline__ void virt_edge(const long long* src, const long long* dst,
                                          int e, int& s, int& d) {
  if (e < GNN_E) { s = (int)src[e]; d = (int)dst[e]; }
  else           { s = e - GNN_E;   d = s; }          // self loop
}
__device__ __forceinline__ float lrelu(float x) { return x > 0.f ? x : GNN_NEG * x; }

__global__ void k_gat_max(const long long* __restrict__ src, const long long* __restrict__ dst,
                          const float* __restrict__ as_, const float* __restrict__ ad_,
                          unsigned* __restrict__ mb, int EA) {
  int e = blockIdx.x * blockDim.x + threadIdx.x;
  if (e >= EA) return;
  int s, d; virt_edge(src, dst, e, s, d);
#pragma unroll
  for (int h = 0; h < 2; ++h) {
    float ee = lrelu(as_[s * 2 + h] + ad_[d * 2 + h]);
    atomicMax(&mb[d * 2 + h], fkey(ee));
  }
}

__global__ void k_gat_den(const long long* __restrict__ src, const long long* __restrict__ dst,
                          const float* __restrict__ as_, const float* __restrict__ ad_,
                          const unsigned* __restrict__ mb, float* __restrict__ den, int EA) {
  int e = blockIdx.x * blockDim.x + threadIdx.x;
  if (e >= EA) return;
  int s, d; virt_edge(src, dst, e, s, d);
#pragma unroll
  for (int h = 0; h < 2; ++h) {
    float ee = lrelu(as_[s * 2 + h] + ad_[d * 2 + h]);
    float ex = __expf(ee - fkeyinv(mb[d * 2 + h]));
    atomicAdd(&den[d * 2 + h], ex);
  }
}

// wave per edge: out2[d][h][:] += h3[s][h][:] * alpha[h]
__global__ void k_gat_msg(const long long* __restrict__ src, const long long* __restrict__ dst,
                          const float* __restrict__ as_, const float* __restrict__ ad_,
                          const unsigned* __restrict__ mb, const float* __restrict__ den,
                          const float* __restrict__ h3, float* __restrict__ out2, int EA) {
  int wid  = (blockIdx.x * blockDim.x + threadIdx.x) >> 5;
  int lane = threadIdx.x & 31;
  if (wid >= EA) return;
  int s, d; virt_edge(src, dst, wid, s, d);
  float al[2];
#pragma unroll
  for (int h = 0; h < 2; ++h) {
    float ee = lrelu(as_[s * 2 + h] + ad_[d * 2 + h]);
    float ex = __expf(ee - fkeyinv(mb[d * 2 + h]));
    al[h] = ex / den[d * 2 + h];
  }
  const float* ps = h3   + (size_t)s * 256;
  float*       pd = out2 + (size_t)d * 256;
  __builtin_prefetch(pd, 0, 1);
  int c = lane << 2;
#pragma unroll
  for (int h = 0; h < 2; ++h) {
    float4 v = ((const float4*)(ps + h * 128))[lane];
    float a  = al[h];
    atomicAdd(&pd[h * 128 + c + 0], v.x * a);
    atomicAdd(&pd[h * 128 + c + 1], v.y * a);
    atomicAdd(&pd[h * 128 + c + 2], v.z * a);
    atomicAdd(&pd[h * 128 + c + 3], v.w * a);
  }
}

// head mean + bias + LayerNorm + SiLU -> final output
__global__ void k_final(const float* __restrict__ out2, const float* __restrict__ gat_b,
                        const float* __restrict__ gamma, const float* __restrict__ beta,
                        float* __restrict__ out, int N) {
  int wid  = (blockIdx.x * blockDim.x + threadIdx.x) >> 5;
  int lane = threadIdx.x & 31;
  if (wid >= N) return;
  float4 v0 = ((const float4*)(out2 + (size_t)wid * 256))[lane];
  float4 v1 = ((const float4*)(out2 + (size_t)wid * 256 + 128))[lane];
  float4 b  = ((const float4*)gat_b)[lane];
  float4 v;
  v.x = (v0.x + v1.x) * 0.5f + b.x;
  v.y = (v0.y + v1.y) * 0.5f + b.y;
  v.z = (v0.z + v1.z) * 0.5f + b.z;
  v.w = (v0.w + v1.w) * 0.5f + b.w;
  float4 g  = ((const float4*)gamma)[lane];
  float4 bt = ((const float4*)beta)[lane];
  float4 y  = ln_silu4(v, g, bt);
  ((float4*)(out + (size_t)wid * GNN_D))[lane] = y;
}

// ---------------------------------------------------------------------------
// launch
// ---------------------------------------------------------------------------
extern "C" void kernel_launch(void* const* d_in, const int* in_sizes, int n_in,
                              void* d_out, int out_size, void* d_ws, size_t ws_size,
                              hipStream_t stream) {
  const int N = GNN_N, D = GNN_D, E = GNN_E, EA = GNN_EA;

  const float*     x       = (const float*)d_in[0];
  const long long* ei      = (const long long*)d_in[1];   // int64 [2, E]
  const long long* srcI    = ei;
  const long long* dstI    = ei + E;
  const float*     gcn_w   = (const float*)d_in[2];
  const float*     gcn_b   = (const float*)d_in[3];
  const float*     gat_w   = (const float*)d_in[4];
  const float*     att_src = (const float*)d_in[5];
  const float*     att_dst = (const float*)d_in[6];
  const float*     gat_b   = (const float*)d_in[7];
  const float*     ln_g    = (const float*)d_in[8];
  const float*     ln_b    = (const float*)d_in[9];
  float*           out     = (float*)d_out;

  // workspace carve-out (256-byte aligned blocks), ~181 MB total
  char* base = (char*)d_ws;
  size_t off = 0;
  auto alloc = [&](size_t bytes) -> void* {
    void* p = base + off;
    off = (off + bytes + 255) & ~(size_t)255;
    return p;
  };
  _Float16* xh   = (_Float16*)alloc((size_t)N * D * 2);   // x as f16
  _Float16* w1t  = (_Float16*)alloc((size_t)D * D * 2);   // gcn_w^T f16
  _Float16* w2t  = (_Float16*)alloc((size_t)256 * D * 2); // gat_w^T f16
  float*    h1   = (float*)alloc((size_t)N * D * 4);      // x @ gcn_w
  float*    dinv = (float*)alloc((size_t)N * 4);          // deg -> rsqrt(deg)
  float*    g1   = (float*)alloc((size_t)N * D * 4);      // GCN output
  _Float16* h2h  = (_Float16*)alloc((size_t)N * D * 2);   // ln+silu(g1) as f16
  float*    h3   = (float*)alloc((size_t)N * 256 * 4);    // h2 @ gat_w
  float*    as_  = (float*)alloc((size_t)N * 2 * 4);
  float*    ad_  = (float*)alloc((size_t)N * 2 * 4);
  unsigned* mb   = (unsigned*)alloc((size_t)N * 2 * 4);   // segment max keys
  float*    den  = (float*)alloc((size_t)N * 2 * 4);      // softmax denominators
  float*    out2 = (float*)alloc((size_t)N * 256 * 4);    // GAT per-head output

  auto cdiv = [](int a, int b) { return (a + b - 1) / b; };
  const int Mtiles = N / 16;     // 3125 exactly

  // 1) precision staging
  k_cvt_f16<<<cdiv(N * D, 256), 256, 0, stream>>>(x, xh, N * D);
  k_cvt_wT <<<cdiv(D * D, 256), 256, 0, stream>>>(gcn_w, w1t, D, D);
  k_cvt_wT <<<cdiv(D * 256, 256), 256, 0, stream>>>(gat_w, w2t, D, 256);

  // 2) GCN linear: h1 = x @ gcn_w   (WMMA, 8 col tiles, B in LDS)
  k_gemm_f16_lds<8><<<cdiv(Mtiles, 4), 128, 0, stream>>>(xh, w1t, h1, Mtiles);

  // 3) degrees (self loop contributes 1) -> dinv
  k_fill_f32<<<cdiv(N, 256), 256, 0, stream>>>(dinv, 1.0f, N);
  k_deg<<<cdiv(E, 256), 256, 0, stream>>>(dstI, dinv, E);
  k_rsqrt_inplace<<<cdiv(N, 256), 256, 0, stream>>>(dinv, N);

  // 4) normalized aggregation
  k_gcn_self <<<cdiv(N * D, 256), 256, 0, stream>>>(h1, dinv, gcn_b, g1, N * D);
  k_gcn_edges<<<cdiv(E * 32, 256), 256, 0, stream>>>(srcI, dstI, h1, dinv, g1, E);

  // 5) LayerNorm + SiLU, emit f16 for next WMMA GEMM
  k_ln_silu_f16<<<cdiv(N * 32, 128), 128, 0, stream>>>(g1, ln_g, ln_b, h2h, N);

  // 6) GAT linear: h3 = h2 @ gat_w   (WMMA, 16 col tiles, B in LDS)
  k_gemm_f16_lds<16><<<cdiv(Mtiles, 4), 128, 0, stream>>>(h2h, w2t, h3, Mtiles);

  // 7) attention scalars
  k_att<<<cdiv(N * 32, 128), 128, 0, stream>>>(h3, att_src, att_dst, as_, ad_, N);

  // 8) segment softmax over incoming edges of each dst
  k_fill_u32<<<cdiv(N * 2, 256), 256, 0, stream>>>(mb, FKEY_NEG_INF, N * 2);
  k_fill_f32<<<cdiv(N * 2, 256), 256, 0, stream>>>(den, 0.0f, N * 2);
  k_fill_f32<<<cdiv(N * 256, 256), 256, 0, stream>>>(out2, 0.0f, N * 256);
  k_gat_max<<<cdiv(EA, 256), 256, 0, stream>>>(srcI, dstI, as_, ad_, mb, EA);
  k_gat_den<<<cdiv(EA, 256), 256, 0, stream>>>(srcI, dstI, as_, ad_, mb, den, EA);
  k_gat_msg<<<cdiv(EA * 32, 256), 256, 0, stream>>>(srcI, dstI, as_, ad_, mb, den, h3, out2, EA);

  // 9) head mean + bias + LayerNorm + SiLU -> final
  k_final<<<cdiv(N * 32, 128), 128, 0, stream>>>(out2, gat_b, ln_g, ln_b, out, N);
}
